// PAMNet_s_13709535609049
// MI455X (gfx1250) — matrix operand order repaired and checked by reading, not compiled, not run on hardware
//
// PAMNet forward for MI455X (gfx1250), compile-only target.
// Core: bf16 WMMA GEMM (v_wmma_f32_16x16x32_bf16); weight tiles staged to LDS
// via TDM (__builtin_amdgcn_tensor_load_to_lds + s_wait_tensorcnt) in the big
// gather GEMM and via per-lane async copies (global_load_async_to_lds_b128 +
// s_wait_asynccnt) in the node GEMMs; LDS-staged A with packed f32->bf16
// conversion; fused gather-concat ingest and fused scatter-atomic epilogue.
#include <hip/hip_runtime.h>
#include <cmath>
#include <cstdint>
#include <cstddef>

#ifndef M_PI
#define M_PI 3.14159265358979323846
#endif

typedef __bf16 v16bf __attribute__((ext_vector_type(16)));
typedef float  v8f   __attribute__((ext_vector_type(8)));
typedef unsigned u32x4 __attribute__((ext_vector_type(4)));
typedef int      i32x4 __attribute__((ext_vector_type(4)));
typedef int      i32x8 __attribute__((ext_vector_type(8)));

#if __has_builtin(__builtin_amdgcn_tensor_load_to_lds)
#define HAVE_TDM 1
#else
#define HAVE_TDM 0
#endif

#define DIM   128
#define NSPH  7
#define NRAD  6
#define CUTL  3.0f
#define CUTG  5.0f

// ---------------------------------------------------------------- helpers
__device__ __forceinline__ unsigned short f2bf(float x) {
  unsigned u = __builtin_bit_cast(unsigned, x);
  u += 0x7FFFu + ((u >> 16) & 1u);               // round-to-nearest-even
  return (unsigned short)(u >> 16);
}
__device__ __forceinline__ unsigned pk2(float lo, float hi) {
  return (unsigned)f2bf(lo) | ((unsigned)f2bf(hi) << 16);
}
__device__ __forceinline__ float siluf(float x) { return x / (1.f + __expf(-x)); }
__device__ __forceinline__ float envf(float d) { // 1/d - 28d^5 + 48d^6 - 21d^7
  float d2 = d * d, d4 = d2 * d2;
  return 1.f / d + d4 * d * (-28.f + d * (48.f - 21.f * d));
}

struct Frag32 { int4 lo, hi; };                  // 32B == 16 bf16 lanes-worth

// ---------------------------------------------------------------- WMMA GEMM
// out[M,128] = silu(A[M,K] @ W[K,128] + bias), K % 32 == 0
//   GATHER      : A row r = concat(h[gi[r]], h[gj[r]], eattr[r])   (K==384)
//   scatter mode: out row r accumulated atomically into out[scat[r]]
struct GemmArgs {
  const float* A;
  const float* h; const int* gi; const int* gj; const float* eattr;
  const unsigned short* W;       // bf16 weights, row-major [K][128] (K padded)
  const float* bias;
  float* out;
  const int* scat;
  int M, K;
};

template <bool GATHER>
__global__ void __launch_bounds__(256)
k_gemm(GemmArgs g) {
  // padded to 40 ushorts/row: stride 80B -> conflict-free ds_load_b128
  __shared__ __align__(16) unsigned short la[128][40];    // A tile [row][k]
  __shared__ __align__(16) unsigned short lbraw[32][128]; // B tile raw [k][col]
  __shared__ __align__(16) unsigned short lb[128][40];    // B tile [col][k]
  const int tid  = threadIdx.x;
  const int wave = tid >> 5;
  const int lane = tid & 31;
  const int gsel = lane >> 4;       // K-half select (wave32 fragment layout)
  const int l16  = lane & 15;
  const int m0   = blockIdx.x * 128;

  // A ingest: each thread owns one half-row (16 contiguous floats)
  const int  arow  = tid >> 1;
  const int  ac0   = (tid & 1) << 4;           // 0 or 16
  const int  agrow = m0 + arow;
  const bool aok   = agrow < g.M;
  const float* as0 = nullptr; const float* as1 = nullptr; const float* as2 = nullptr;
  if (GATHER) {
    if (aok) {
      as0 = g.h + (size_t)g.gi[agrow] * DIM;
      as1 = g.h + (size_t)g.gj[agrow] * DIM;
      as2 = g.eattr + (size_t)agrow * DIM;
    }
  }
  // B async ingest (non-TDM path): 16 contiguous bf16 of one k-row per thread
  const int bkr = tid >> 3;                    // 0..31
  const int bc0 = (tid & 7) << 4;              // 0..112
  // LDS byte offset == low 32 bits of the flat shared-pointer (aperture map)
  const unsigned ldsb    = (unsigned)(uintptr_t)&lbraw[bkr][bc0];
  const unsigned ldsbase = (unsigned)(uintptr_t)&lbraw[0][0];

  v8f acc[8] = {};
  const int nk = g.K >> 5;
  for (int s = 0; s < nk; ++s) {
    const int kk = s << 5;
    // ---- A tile: 16 f32 -> packed bf16 -> 2x ds_store_b128
    {
      uint4 p0, p1;
      p0.x = p0.y = p0.z = p0.w = 0u; p1 = p0;
      if (aok) {
        const float* base;
        if (GATHER) {
          base = (kk < DIM) ? (as0 + kk) : (kk < 2 * DIM) ? (as1 + kk - DIM)
                                                          : (as2 + kk - 2 * DIM);
        } else {
          base = g.A + (size_t)agrow * g.K + kk;
        }
        base += ac0;
        float4 f0 = *(const float4*)(base + 0);
        float4 f1 = *(const float4*)(base + 4);
        float4 f2 = *(const float4*)(base + 8);
        float4 f3 = *(const float4*)(base + 12);
        p0.x = pk2(f0.x, f0.y); p0.y = pk2(f0.z, f0.w);
        p0.z = pk2(f1.x, f1.y); p0.w = pk2(f1.z, f1.w);
        p1.x = pk2(f2.x, f2.y); p1.y = pk2(f2.z, f2.w);
        p1.z = pk2(f3.x, f3.y); p1.w = pk2(f3.z, f3.w);
        if (ac0 == 0 && s + 1 < nk)            // gfx1250 global_prefetch_b8
          __builtin_prefetch(base + 32, 0, 1);
      }
      *(uint4*)&la[arow][ac0]     = p0;
      *(uint4*)&la[arow][ac0 + 8] = p1;
    }
    // ---- B tile staging: TDM DMA (gather GEMM) or per-lane async copies
#if HAVE_TDM
    if (GATHER) {
      if (wave == 0) {
        // Tensor DMA: 2D tile 128(cols) x 32(k-rows), bf16, contiguous rows.
        unsigned long long ga = (unsigned long long)(uintptr_t)(g.W + (size_t)kk * DIM);
        u32x4 g0;
        g0[0] = 1u;                                             // count=1, user D#
        g0[1] = ldsbase;                                        // lds_addr
        g0[2] = (unsigned)ga;                                   // global_addr[31:0]
        g0[3] = (unsigned)((ga >> 32) & 0x01FFFFFFu) | (2u << 30); // [56:32]|type=2
        i32x8 g1;
        g1[0] = 0x00010000;          // workgroup_mask=0, data_size=1 (2 bytes)
        g1[1] = (int)(128u << 16);   // tensor_dim0 = 128 (low16 at [63:48])
        g1[2] = (int)(32u << 16);    // tensor_dim1 = 32  (low16 at [95:80])
        g1[3] = (int)(128u << 16);   // tile_dim0 = 128   ([127:112])
        g1[4] = 32;                  // tile_dim1 = 32    ([143:128])
        g1[5] = 128;                 // tensor_dim0_stride = 128 ([191:160])
        g1[6] = 0; g1[7] = 0;
        i32x4 z4 = {};               // groups 2/3 unused (2D tensor)
#if defined(__clang_major__) && (__clang_major__ >= 23)
        i32x8 z8 = {};
        __builtin_amdgcn_tensor_load_to_lds(g0, g1, z4, z4, z8, 0);
#else
        __builtin_amdgcn_tensor_load_to_lds(g0, g1, z4, z4, 0);
#endif
        __builtin_amdgcn_s_wait_tensorcnt(0);
      }
    } else
#endif
    {
      const unsigned short* src = g.W + (size_t)(kk + bkr) * DIM + bc0;
      unsigned long long ga = (unsigned long long)(uintptr_t)src;
      asm volatile("global_load_async_to_lds_b128 %0, %1, off"
                   :: "v"(ldsb), "v"(ga) : "memory");
      asm volatile("global_load_async_to_lds_b128 %0, %1, off offset:16"
                   :: "v"(ldsb), "v"(ga) : "memory");
      asm volatile("s_wait_asynccnt 0x0" ::: "memory");
    }
    __syncthreads();
    // ---- transpose B raw [k][col] -> padded [col][k], packed k-pair stores
    {
      const int rp = (tid >> 4) << 1;        // even k-row of the pair
      const int c0 = (tid & 15) << 3;        // 8 columns
      int4 r0 = *(const int4*)&lbraw[rp][c0];
      int4 r1 = *(const int4*)&lbraw[rp + 1][c0];
      unsigned short u0[8], u1[8];
      *(int4*)u0 = r0; *(int4*)u1 = r1;
      #pragma unroll
      for (int u = 0; u < 8; ++u) {
        unsigned pv = (unsigned)u0[u] | ((unsigned)u1[u] << 16);
        *(unsigned*)&lb[c0 + u][rp] = pv;    // two adjacent k, 4B aligned
      }
    }
    __syncthreads();
    // ---- fragments + WMMA
    Frag32 fa;
    const int frow = wave * 16 + l16;
    fa.lo = *(const int4*)&la[frow][gsel * 8];
    fa.hi = *(const int4*)&la[frow][16 + gsel * 8];
    v16bf a = __builtin_bit_cast(v16bf, fa);
    #pragma unroll
    for (int j = 0; j < 8; ++j) {
      Frag32 fb;
      const int col = j * 16 + l16;
      fb.lo = *(const int4*)&lb[col][gsel * 8];
      fb.hi = *(const int4*)&lb[col][16 + gsel * 8];
      v16bf b = __builtin_bit_cast(v16bf, fb);
      acc[j] = __builtin_amdgcn_wmma_f32_16x16x32_bf16(false, a, false, b,
                                                       (short)0, acc[j], false, false);
    }
    __syncthreads();
  }

  // epilogue: C/D layout -> VGPR r: row = 8*gsel + r, col = lane&15
  #pragma unroll
  for (int j = 0; j < 8; ++j) {
    const int col = j * 16 + l16;
    const float bia = g.bias[col];
    #pragma unroll
    for (int r = 0; r < 8; ++r) {
      int grow = m0 + wave * 16 + gsel * 8 + r;
      if (grow >= g.M) continue;
      float v = siluf(acc[j][r] + bia);
      if (g.scat) atomicAdd(&g.out[(size_t)g.scat[grow] * DIM + col], v);
      else        g.out[(size_t)grow * DIM + col] = v;
    }
  }
}

// ---------------------------------------------------------------- small kernels
__global__ void k_embed(const float* emb, const int* xraw, float* x, int N) {
  int i = blockIdx.x * blockDim.x + threadIdx.x;
  if (i >= N * DIM) return;
  x[i] = emb[(size_t)xraw[i >> 7] * DIM + (i & 127)];
}
__global__ void k_fill(float* p, int n, float v) {
  int i = blockIdx.x * blockDim.x + threadIdx.x;
  if (i < n) p[i] = v;
}
__global__ void k_copy(float* d, const float* s, int n) {
  int i = blockIdx.x * blockDim.x + threadIdx.x;
  if (i < n) d[i] = s[i];
}
__global__ void k_add2(float* d, const float* a, const float* b, int n) {
  int i = blockIdx.x * blockDim.x + threadIdx.x;
  if (i < n) d[i] = a[i] + b[i];
}
__global__ void k_add3(float* d, const float* a, const float* b, const float* c, int n) {
  int i = blockIdx.x * blockDim.x + threadIdx.x;
  if (i < n) d[i] = a[i] + b[i] + c[i];
}
// f32 -> bf16 with trailing-row zero padding (rows beyond nsrc are 0)
__global__ void k_convert_pad(const float* s, unsigned short* d, int nsrc, int ntot) {
  int i = blockIdx.x * blockDim.x + threadIdx.x;
  if (i < ntot) d[i] = (i < nsrc) ? f2bf(s[i]) : (unsigned short)0;
}
__global__ void k_edge_rbf(const float* pos, const int* jidx, const int* iidx, int E,
                           const float* freq, float cutoff, float* raw, float* dist_out) {
  int e = blockIdx.x * blockDim.x + threadIdx.x;
  if (e >= E) return;
  int j = jidx[e], i = iidx[e];
  float dx = pos[3*i] - pos[3*j], dy = pos[3*i+1] - pos[3*j+1], dz = pos[3*i+2] - pos[3*j+2];
  float dist = sqrtf(dx*dx + dy*dy + dz*dz);
  if (dist_out) dist_out[e] = dist;
  float d = dist / cutoff;
  float en = envf(d);
  for (int k = 0; k < 32; ++k)   // stride 32, zero-padded cols 16..31
    raw[(size_t)e * 32 + k] = (k < 16) ? en * __sinf(freq[k] * d) : 0.f;
}
struct SbfC { float z[NSPH*NRAD]; float nrm[NSPH*NRAD]; float sph[NSPH]; };
__device__ __forceinline__ float jl_dev(int l, float x) {
  float j0 = __sinf(x) / x;
  if (l == 0) return j0;
  float j1 = __sinf(x) / (x * x) - __cosf(x) / x;
  float jm = j0, jc = j1;
  for (int n = 1; n < l; ++n) { float t = (2.f*n + 1.f) / x * jc - jm; jm = jc; jc = t; }
  return jc;
}
__global__ void k_edge_rad(const float* dist, int E, SbfC C, float* rad) {
  int e = blockIdx.x * blockDim.x + threadIdx.x;
  if (e >= E) return;
  float d = dist[e] / CUTL;
  float en = envf(d);
  for (int l = 0; l < NSPH; ++l)
    for (int k = 0; k < NRAD; ++k) {
      float x = C.z[l*NRAD + k] * d;
      rad[(size_t)e*NSPH*NRAD + l*NRAD + k] = en * C.nrm[l*NRAD + k] * jl_dev(l, x);
    }
}
__global__ void k_pair_sbf(const float* pos, const int* ii, const int* ij1, const int* ij2,
                           const int* ijj, int T, const float* rad, SbfC C, float* raw) {
  int t = blockIdx.x * blockDim.x + threadIdx.x;
  if (t >= T) return;
  int a = ii[t], b = ij1[t], c = ij2[t];
  float v1x = pos[3*b]-pos[3*a], v1y = pos[3*b+1]-pos[3*a+1], v1z = pos[3*b+2]-pos[3*a+2];
  float v2x = pos[3*c]-pos[3*b], v2y = pos[3*c+1]-pos[3*b+1], v2z = pos[3*c+2]-pos[3*b+2];
  float dt = v1x*v2x + v1y*v2y + v1z*v2z;
  float cx = v1y*v2z - v1z*v2y, cy = v1z*v2x - v1x*v2z, cz = v1x*v2y - v1y*v2x;
  float ss = cx*cx + cy*cy + cz*cz;
  float bn = ss > 0.f ? sqrtf(ss) : 0.f;
  float cth = __cosf(atan2f(bn, dt));
  float P[NSPH]; P[0] = 1.f; P[1] = cth;
  for (int l = 1; l < NSPH - 1; ++l)
    P[l+1] = ((2.f*l + 1.f) * cth * P[l] - (float)l * P[l-1]) / (float)(l + 1);
  int e = ijj[t];
  for (int l = 0; l < NSPH; ++l) {
    float cb = P[l] * C.sph[l];
    for (int k = 0; k < NRAD; ++k)
      raw[(size_t)t*64 + l*NRAD + k] = rad[(size_t)e*NSPH*NRAD + l*NRAD + k] * cb;
  }
  for (int k = NSPH * NRAD; k < 64; ++k) raw[(size_t)t*64 + k] = 0.f;  // pad 42..63
}
__global__ void k_pair_mix(const float* m, const float* sbf, const int* ijj, const int* iji,
                           int T, float* m2) {
  int i = blockIdx.x * blockDim.x + threadIdx.x;
  if (i >= T * DIM) return;
  int t = i >> 7, c = i & 127;
  atomicAdd(&m2[(size_t)iji[t] * DIM + c], m[(size_t)ijj[t] * DIM + c] * sbf[i]);
}
__global__ void k_scatter_add(float* dst, const float* src, const int* idx, int E) {
  int i = blockIdx.x * blockDim.x + threadIdx.x;
  if (i >= E * DIM) return;
  int e = i >> 7, c = i & 127;
  atomicAdd(&dst[(size_t)idx[e] * DIM + c], src[i]);
}
__global__ void __launch_bounds__(128)
k_heads(const float* oh, const float* Wo, const float* bo, const float* Wa, const float* ba,
        float* ov, float* av, int N) {
  __shared__ float s1[128], s2[128];
  int n = blockIdx.x, t = threadIdx.x;
  float v = oh[(size_t)n * DIM + t];
  s1[t] = v * Wo[t]; s2[t] = v * Wa[t];
  __syncthreads();
  for (int o = 64; o > 0; o >>= 1) {
    if (t < o) { s1[t] += s1[t + o]; s2[t] += s2[t + o]; }
    __syncthreads();
  }
  if (t == 0) { ov[n] = s1[0] + bo[0]; av[n] = s2[0] + ba[0]; }
}
__global__ void k_finalize(const float* og, const float* ol, const float* ag, const float* al,
                           const int* batch, int N, float* out) {
  int n = blockIdx.x * blockDim.x + threadIdx.x;
  if (n >= N) return;
  float s = 0.f;
  for (int l = 0; l < 3; ++l) {
    float g = ag[l*N + n], lo = al[l*N + n];
    g = g > 0.f ? g : 0.2f * g;
    lo = lo > 0.f ? lo : 0.2f * lo;
    float mx = fmaxf(g, lo);
    float wg = __expf(g - mx), wl = __expf(lo - mx);
    s += (wg * og[l*N + n] + wl * ol[l*N + n]) / (wg + wl);
  }
  atomicAdd(&out[batch[n]], s);
}

// ---------------------------------------------------------------- host: bessel zeros
static double np_jl(int l, double x) {
  double j0 = sin(x) / x;
  if (l == 0) return j0;
  double j1 = sin(x) / (x * x) - cos(x) / x;
  double jm = j0, jc = j1;
  for (int n = 1; n < l; ++n) { double t = (2.0*n + 1.0) / x * jc - jm; jm = jc; jc = t; }
  return jc;
}
static void build_sbfc(SbfC& C) {
  const int NP = NRAD + NSPH - 1; // 12
  double zerosj[NSPH][NRAD];
  for (int k = 0; k < NRAD; ++k) zerosj[0][k] = (k + 1) * M_PI;
  double points[NP], racines[NP];
  for (int j = 0; j < NP; ++j) { points[j] = (j + 1) * M_PI; racines[j] = 0.0; }
  for (int l = 1; l < NSPH; ++l) {
    for (int j = 0; j < NP - l; ++j) {
      double a = points[j], b = points[j + 1], fa = np_jl(l, a);
      for (int it = 0; it < 80; ++it) {
        double m = 0.5 * (a + b), fm = np_jl(l, m);
        if (fa * fm <= 0) b = m; else { a = m; fa = fm; }
      }
      racines[j] = 0.5 * (a + b);
    }
    for (int j = 0; j < NP; ++j) points[j] = racines[j];
    for (int k = 0; k < NRAD; ++k) zerosj[l][k] = racines[k];
  }
  for (int l = 0; l < NSPH; ++l) {
    C.sph[l] = (float)sqrt((2.0*l + 1.0) / (4.0 * M_PI));
    for (int k = 0; k < NRAD; ++k) {
      C.z[l*NRAD + k] = (float)zerosj[l][k];
      double v = np_jl(l + 1, zerosj[l][k]);
      C.nrm[l*NRAD + k] = (float)(1.0 / sqrt(0.5 * v * v));
    }
  }
}

// ---------------------------------------------------------------- launch
// Param leaf order: jax pytree flatten (dict keys sorted).
//  0 embeddings | 1 freq_g | 2 freq_l | 3..74 global[0..2] | 75..146 local[0..2]
//  147/148 mlp_rbf_g W/b | 149/150 mlp_rbf_l W/b | 151/152 mlp_sbf W/b
//  per layer (+): 0 att.W 1 att.b 2 m.W 3 m.b 4 out.W 5 out.b 6 x1.W 7 x1.b
//                 8 x2.W 9 x2.b 10 head.W 11 head.b 12..15 res1 16..19 res2 20..23 res3
//  153 pos | 154 x_raw | 155 batch | 156 eil | 157 eig | 158..162 pair idx | 163 n_graphs
static inline int cdiv(long long a, long long b) { return (int)((a + b - 1) / b); }

extern "C" void kernel_launch(void* const* d_in, const int* in_sizes, int n_in,
                              void* d_out, int out_size, void* d_ws, size_t ws_size,
                              hipStream_t stream) {
  (void)n_in; (void)ws_size;
  auto PF = [&](int i) { return (const float*)d_in[i]; };
  auto PI = [&](int i) { return (const int*)d_in[i]; };

  const int N  = in_sizes[154];
  const int El = in_sizes[156] / 2;
  const int Eg = in_sizes[157] / 2;
  const int T  = in_sizes[158];
  const int maxE = El > Eg ? El : Eg;

  char* wsb = (char*)d_ws;
  size_t off = 0;
  auto alloc = [&](size_t bytes) -> void* {
    size_t o = off; off = (o + bytes + 255) & ~(size_t)255; return wsb + o;
  };
  // bf16 weight arena: 6 stages * (384*128 + 9*128*128) + padded rbf_g/rbf_l/sbf
  const size_t STG = 196608;
  const size_t OFF_RBFG = 6 * STG;             // 32x128
  const size_t OFF_RBFL = OFF_RBFG + 4096;     // 32x128
  const size_t OFF_SBF  = OFF_RBFL + 4096;     // 64x128
  unsigned short* wbf = (unsigned short*)alloc((OFF_SBF + 8192) * 2);
  float* xA   = (float*)alloc((size_t)N * DIM * 4);
  float* xB   = (float*)alloc((size_t)N * DIM * 4);
  float* t1   = (float*)alloc((size_t)N * DIM * 4);
  float* t2   = (float*)alloc((size_t)N * DIM * 4);
  float* agg  = (float*)alloc((size_t)N * DIM * 4);
  float* h    = (float*)alloc((size_t)N * DIM * 4);
  float* oh   = (float*)alloc((size_t)N * DIM * 4);
  float* rraw = (float*)alloc((size_t)maxE * 32 * 4);   // K padded 16 -> 32
  float* rbfg = (float*)alloc((size_t)Eg * DIM * 4);
  float* rbfl = (float*)alloc((size_t)El * DIM * 4);
  float* dstl = (float*)alloc((size_t)El * 4);
  float* rad  = (float*)alloc((size_t)El * 42 * 4);
  float* sraw = (float*)alloc((size_t)T * 64 * 4);      // K padded 42 -> 64
  float* sbff = (float*)alloc((size_t)T * DIM * 4);
  float* mbuf = (float*)alloc((size_t)El * DIM * 4);
  float* m2   = (float*)alloc((size_t)El * DIM * 4);
  float* outg = (float*)alloc((size_t)3 * N * 4);
  float* outl = (float*)alloc((size_t)3 * N * 4);
  float* attg = (float*)alloc((size_t)3 * N * 4);
  float* attl = (float*)alloc((size_t)3 * N * 4);

  auto conv = [&](const float* src, size_t dstoff, int n) {
    k_convert_pad<<<cdiv(n, 256), 256, 0, stream>>>(src, wbf + dstoff, n, n);
  };
  auto convpad = [&](const float* src, size_t dstoff, int nsrc, int ntot) {
    k_convert_pad<<<cdiv(ntot, 256), 256, 0, stream>>>(src, wbf + dstoff, nsrc, ntot);
  };
  auto fill = [&](float* p, int n, float v) {
    if (n > 0) k_fill<<<cdiv(n, 256), 256, 0, stream>>>(p, n, v);
  };
  auto G = [&](const float* A, const unsigned short* W, const float* bias, float* out,
               int M, int K, const int* scat = nullptr, const float* hh = nullptr,
               const int* gi = nullptr, const int* gj = nullptr, const float* ea = nullptr) {
    if (M <= 0) return;
    GemmArgs a; a.A = A; a.h = hh; a.gi = gi; a.gj = gj; a.eattr = ea;
    a.W = W; a.bias = bias; a.out = out; a.scat = scat; a.M = M; a.K = K;
    if (hh) k_gemm<true ><<<dim3(cdiv(M, 128)), dim3(256), 0, stream>>>(a);
    else    k_gemm<false><<<dim3(cdiv(M, 128)), dim3(256), 0, stream>>>(a);
  };

  // --- convert all GEMM weights to bf16 (deterministic each call) ---
  for (int s = 0; s < 6; ++s) {
    int l = s >> 1; bool loc = s & 1;
    int pb = loc ? (75 + 24 * l) : (3 + 24 * l);
    size_t wb = (size_t)s * STG;
    conv(PF(pb + 2),  wb + 0,      49152);  // mlp_m
    conv(PF(pb + 4),  wb + 49152,  16384);  // mlp_out
    conv(PF(pb + 6),  wb + 65536,  16384);  // mlp_x1
    conv(PF(pb + 8),  wb + 81920,  16384);  // mlp_x2
    conv(PF(pb + 12), wb + 98304,  16384);  // res1a
    conv(PF(pb + 14), wb + 114688, 16384);  // res1b
    conv(PF(pb + 16), wb + 131072, 16384);  // res2a
    conv(PF(pb + 18), wb + 147456, 16384);  // res2b
    conv(PF(pb + 20), wb + 163840, 16384);  // res3a
    conv(PF(pb + 22), wb + 180224, 16384);  // res3b
  }
  convpad(PF(147), OFF_RBFG, 2048, 4096);  // rbf_g 16x128 -> 32x128
  convpad(PF(149), OFF_RBFL, 2048, 4096);  // rbf_l 16x128 -> 32x128
  convpad(PF(151), OFF_SBF,  5376, 8192);  // sbf  42x128 -> 64x128

  // --- features ---
  const int* jg = PI(157); const int* ig = jg + Eg;
  const int* jl = PI(156); const int* il = jl + El;
  SbfC C; build_sbfc(C);

  k_embed<<<cdiv((long long)N * DIM, 256), 256, 0, stream>>>(PF(0), PI(154), xA, N);
  k_edge_rbf<<<cdiv(Eg, 128), 128, 0, stream>>>(PF(153), jg, ig, Eg, PF(1), CUTG, rraw, nullptr);
  G(rraw, wbf + OFF_RBFG, PF(148), rbfg, Eg, 32);
  k_edge_rbf<<<cdiv(El, 128), 128, 0, stream>>>(PF(153), jl, il, El, PF(2), CUTL, rraw, dstl);
  G(rraw, wbf + OFF_RBFL, PF(150), rbfl, El, 32);
  k_edge_rad<<<cdiv(El, 128), 128, 0, stream>>>(dstl, El, C, rad);
  k_pair_sbf<<<cdiv(T, 128), 128, 0, stream>>>(PF(153), PI(158), PI(159), PI(160), PI(161),
                                               T, rad, C, sraw);
  G(sraw, wbf + OFF_SBF, PF(152), sbff, T, 64);

  // --- 6 message-passing stages: (global, local) x 3 layers ---
  for (int s = 0; s < 6; ++s) {
    int l = s >> 1; bool loc = s & 1;
    int pb = loc ? (75 + 24 * l) : (3 + 24 * l);
    const unsigned short* wb = wbf + (size_t)s * STG;
    const unsigned short *Wm = wb, *Wout = wb + 49152, *Wx1 = wb + 65536, *Wx2 = wb + 81920;
    const unsigned short *R1a = wb + 98304, *R1b = wb + 114688, *R2a = wb + 131072,
                         *R2b = wb + 147456, *R3a = wb + 163840, *R3b = wb + 180224;

    G(xA, Wx1, PF(pb + 7), h, N, DIM);                         // h = silu(x1(x))
    fill(agg, N * DIM, 0.f);
    if (!loc) {
      // message GEMM, K=384 gather-concat, epilogue scatter-atomic into agg
      G(nullptr, Wm, PF(pb + 3), agg, Eg, 3 * DIM, ig, h, ig, jg, rbfg);
    } else {
      G(nullptr, Wm, PF(pb + 3), mbuf, El, 3 * DIM, nullptr, h, il, jl, rbfl);
      k_copy<<<cdiv((long long)El * DIM, 256), 256, 0, stream>>>(m2, mbuf, El * DIM);
      k_pair_mix<<<cdiv((long long)T * DIM, 256), 256, 0, stream>>>(mbuf, sbff, PI(161),
                                                                    PI(162), T, m2);
      k_scatter_add<<<cdiv((long long)El * DIM, 256), 256, 0, stream>>>(agg, m2, il, El);
    }
    // x = res1(agg) + res_x ; then x2 ; then res2 ; res3
    G(agg, R1a, PF(pb + 13), t1, N, DIM);
    G(t1,  R1b, PF(pb + 15), t2, N, DIM);
    k_add3<<<cdiv((long long)N * DIM, 256), 256, 0, stream>>>(t1, agg, t2, xA, N * DIM);
    G(t1,  Wx2, PF(pb + 9), xB, N, DIM);
    G(xB,  R2a, PF(pb + 17), t2, N, DIM);
    G(t2,  R2b, PF(pb + 19), t1, N, DIM);
    k_add2<<<cdiv((long long)N * DIM, 256), 256, 0, stream>>>(xB, xB, t1, N * DIM);
    G(xB,  R3a, PF(pb + 21), t2, N, DIM);
    G(t2,  R3b, PF(pb + 23), t1, N, DIM);
    k_add2<<<cdiv((long long)N * DIM, 256), 256, 0, stream>>>(xB, xB, t1, N * DIM);
    G(xB,  Wout, PF(pb + 5), oh, N, DIM);                      // oh = silu(mlp_out(x))
    float* ov = (loc ? outl : outg) + (size_t)l * N;
    float* av = (loc ? attl : attg) + (size_t)l * N;
    k_heads<<<dim3(N), dim3(128), 0, stream>>>(oh, PF(pb + 10), PF(pb + 11),
                                               PF(pb + 0), PF(pb + 1), ov, av, N);
    float* tp = xA; xA = xB; xB = tp;
  }

  fill((float*)d_out, out_size, 0.f);
  k_finalize<<<cdiv(N, 256), 256, 0, stream>>>(outg, outl, attg, attl, PI(155), N,
                                               (float*)d_out);
}
